// PreconditionerSparseUNet_8753143349639
// MI455X (gfx1250) — compile-verified, matrix operand order checked by reading, not run on hardware
//
#include <hip/hip_runtime.h>
#include <hip/hip_bf16.h>
#include <math.h>

// ---------------------------------------------------------------------------
// PreconditionerSparseUNet on MI455X (gfx1250).
//  * every conv / tconv layer = implicit GEMM on v_wmma_f32_16x16x32_f16
//  * f16 activations in workspace, f32 accumulate, fused bias+LeakyReLU+skip
//  * weights pre-transposed to [Cout][KPAD] so the per-block LDS stage is a
//    dense 2D tile -> loaded with the Tensor Data Mover (tensor_load_to_lds)
//  * A fragments: two global_load_b128 per lane per K-step (contiguous K runs)
//  * B fragments: two ds_load_b128 per accumulator per K-step
// ---------------------------------------------------------------------------

typedef __attribute__((ext_vector_type(16))) _Float16 v16h;
typedef __attribute__((ext_vector_type(8)))  _Float16 h8;
typedef __attribute__((ext_vector_type(8)))  float    v8f;
typedef __attribute__((ext_vector_type(4)))  unsigned int u32x4;
typedef __attribute__((ext_vector_type(8)))  int i32x8;
typedef __attribute__((ext_vector_type(4)))  int i32x4;

#define BATCH 8
#define NSZ   512

#define TDM_OK 0
#if defined(__has_builtin)
#if __has_builtin(__builtin_amdgcn_tensor_load_to_lds) && __has_builtin(__builtin_amdgcn_s_wait_tensorcnt)
#undef TDM_OK
#define TDM_OK 1
#endif
#endif

// --------------------------- f32 -> f16 cast -------------------------------
__global__ void cast_f32_to_f16_kernel(const float* __restrict__ src,
                                       _Float16* __restrict__ dst, int n) {
    int i = blockIdx.x * 256 + threadIdx.x;
    if (i < n) dst[i] = (_Float16)src[i];
}

// ---------- weight prep: HWIO f32 -> transposed padded f16 [Cout][KPAD] ----
__global__ void prep_w_kernel(const float* __restrict__ w, _Float16* __restrict__ wT,
                              int Cout, int Kreal, int Kpad) {
    int i = blockIdx.x * 256 + threadIdx.x;
    if (i >= Cout * Kpad) return;
    int co = i / Kpad;
    int k  = i - co * Kpad;
    wT[i] = (k < Kreal) ? (_Float16)w[k * Cout + co] : (_Float16)0.f;
}

__device__ inline h8 h8zero() {
    h8 z;
#pragma unroll
    for (int t = 0; t < 8; ++t) z[t] = (_Float16)0.f;
    return z;
}

// ------------------------ implicit-GEMM conv (WMMA) ------------------------
// Block: 256 threads = 8 wave32; each wave owns 16 output pixels x (16*NT)
// output channels.  grid.x: pixel tiles (128 px / block); grid.y: Cout/(16*NT).
template <int CINSHIFT, int STRIDE, int TRANSPOSED, int NT>
__global__ __launch_bounds__(256, 1)
void conv_wmma_kernel(const _Float16* __restrict__ in,
                      const _Float16* __restrict__ wT,   // [Cout][KPAD] f16
                      const float*    __restrict__ bias,
                      const _Float16* __restrict__ skip, // may be null
                      _Float16*       __restrict__ out,
                      int hinShift, int wShift, int Cout) {
    constexpr int CIN   = 1 << CINSHIFT;
    constexpr int KREAL = 9 * CIN;
    constexpr int KPAD  = (KREAL + 31) & ~31;
    constexpr int KSTEPS = KPAD >> 5;
    constexpr int NCOLS = 16 * NT;

    __shared__ _Float16 lds_w[NCOLS * KPAD];

    const int Hin  = 1 << hinShift;
    const int Hout = 1 << wShift;
    const int npix = BATCH << (2 * wShift);
    const int tid  = threadIdx.x;
    const int ntBase = blockIdx.y * NCOLS;
    const _Float16* wsrc = wT + (size_t)ntBase * KPAD;

    // ---- stage this column-tile's weights (NCOLS x KPAD, dense) into LDS ---
#if TDM_OK
    if (tid == 0) {
        unsigned long long ga = (unsigned long long)(uintptr_t)wsrc;
        unsigned lds_lo = (unsigned)(unsigned long long)(uintptr_t)(&lds_w[0]);
        // D# group0: count=1, lds_addr, global_addr[56:0], type=2
        u32x4 g0 = { 1u, lds_lo, (unsigned)ga,
                     (unsigned)((ga >> 32) & 0x01ffffffu) | 0x80000000u };
        // D# group1: data_size=2B, tensor_dim0=KPAD, tensor_dim1=NCOLS,
        //            tile_dim0=KPAD, tile_dim1=NCOLS, dim0_stride=KPAD
        i32x8 g1 = { (int)0x10000u,
                     (int)(((unsigned)KPAD & 0xffffu) << 16),
                     (int)((((unsigned)KPAD >> 16) & 0xffffu) |
                           (((unsigned)NCOLS & 0xffffu) << 16)),
                     (int)(((unsigned)NCOLS >> 16) | ((unsigned)KPAD << 16)),
                     (int)(unsigned)NCOLS,
                     (int)(unsigned)KPAD,
                     0, 0 };
        i32x4 gz = {0, 0, 0, 0};
#if __clang_major__ >= 23
        i32x8 gz8 = {0, 0, 0, 0, 0, 0, 0, 0};
        __builtin_amdgcn_tensor_load_to_lds(g0, g1, gz, gz, gz8, 0);
#else
        __builtin_amdgcn_tensor_load_to_lds(g0, g1, gz, gz, 0);
#endif
    }
    __builtin_amdgcn_s_wait_tensorcnt(0);
    __syncthreads();
#else
    for (int i = tid; i < NCOLS * KPAD; i += 256) lds_w[i] = wsrc[i];
    __syncthreads();
#endif

    const int wave = tid >> 5;
    const int lane = tid & 31;
    const int tile = blockIdx.x * 8 + wave;
    const int hi   = lane >> 4;
    const int khalf = hi << 3;               // 0 or 8

    const int pix = tile * 16 + (lane & 15); // A-matrix row (M)
    const bool pv = pix < npix;
    const int bb  = pix >> (2 * wShift);
    const int oy  = (pix >> wShift) & (Hout - 1);
    const int ox  = pix & (Hout - 1);

    // Load one contiguous run of 8 K-values of the A matrix for K start kr.
    auto load_runA = [&](int kr) -> h8 {
        if (!pv || kr >= KREAL) return h8zero();
        const int tap = kr >> CINSHIFT;      // one tap per 8-run (CIN >= 8)
        const int ky  = tap >= 6 ? 2 : (tap >= 3 ? 1 : 0);
        const int kx  = tap - ky * 3;
        int iy, ix;
        bool v;
        if constexpr (!TRANSPOSED) {
            iy = oy * STRIDE + ky - 1;
            ix = ox * STRIDE + kx - 1;
            v  = ((unsigned)iy < (unsigned)Hin) & ((unsigned)ix < (unsigned)Hin);
        } else {  // lhs_dilation=2, pad(1,2): tap valid only on parity match
            const int ty = oy + 1 - ky;
            const int tx = ox + 1 - kx;
            v  = (ty >= 0) & (tx >= 0) & !(ty & 1) & !(tx & 1);
            iy = ty >> 1;
            ix = tx >> 1;
            v  = v & (iy < Hin) & (ix < Hin);
        }
        if (!v) return h8zero();
        size_t base = ((((size_t)bb << hinShift) + iy) << hinShift) + ix;
        base = (base << CINSHIFT) + (kr & (CIN - 1));
        return *(const h8*)(in + base);      // global_load_b128
    };

    v8f acc[NT];
#pragma unroll
    for (int j = 0; j < NT; ++j)
#pragma unroll
        for (int r = 0; r < 8; ++r) acc[j][r] = 0.f;

    for (int ks = 0; ks < KSTEPS; ++ks) {
        const int k0 = ks << 5;
        h8 a0, a1;
        if constexpr (CINSHIFT >= 4) {
            a0 = load_runA(k0 + khalf);
            a1 = load_runA(k0 + 16 + khalf);
        } else {  // Cin==1 (enc1): scalar im2col gather
            a0 = h8zero(); a1 = h8zero();
#pragma unroll
            for (int t = 0; t < 8; ++t) {
                const int kr0 = k0 + khalf + t;
                const int kr1 = k0 + 16 + khalf + t;
                if (pv && kr0 < KREAL) {
                    const int ky = kr0 >= 6 ? 2 : (kr0 >= 3 ? 1 : 0);
                    const int kx = kr0 - ky * 3;
                    const int iy = oy * STRIDE + ky - 1;
                    const int ix = ox * STRIDE + kx - 1;
                    if (((unsigned)iy < (unsigned)Hin) & ((unsigned)ix < (unsigned)Hin))
                        a0[t] = in[((((size_t)bb << hinShift) + iy) << hinShift) + ix];
                }
                if (pv && kr1 < KREAL) {
                    const int ky = kr1 >= 6 ? 2 : (kr1 >= 3 ? 1 : 0);
                    const int kx = kr1 - ky * 3;
                    const int iy = oy * STRIDE + ky - 1;
                    const int ix = ox * STRIDE + kx - 1;
                    if (((unsigned)iy < (unsigned)Hin) & ((unsigned)ix < (unsigned)Hin))
                        a1[t] = in[((((size_t)bb << hinShift) + iy) << hinShift) + ix];
                }
            }
        }
        v16h amat = __builtin_shufflevector(a0, a1, 0, 1, 2, 3, 4, 5, 6, 7,
                                            8, 9, 10, 11, 12, 13, 14, 15);
        if (ks + 1 < KSTEPS)  // gfx1250 global_prefetch_b8 on next K-slab
            __builtin_prefetch((const void*)(in +
                ((((size_t)bb << hinShift) + oy) << hinShift << CINSHIFT) + ((ks + 1) << 5)),
                0, 1);
#pragma unroll
        for (int j = 0; j < NT; ++j) {
            const _Float16* col = lds_w + (size_t)(j * 16 + (lane & 15)) * KPAD;
            h8 b0 = *(const h8*)(col + k0 + khalf);        // ds_load_b128
            h8 b1 = *(const h8*)(col + k0 + 16 + khalf);   // ds_load_b128
            v16h bmat = __builtin_shufflevector(b0, b1, 0, 1, 2, 3, 4, 5, 6, 7,
                                                8, 9, 10, 11, 12, 13, 14, 15);
            acc[j] = __builtin_amdgcn_wmma_f32_16x16x32_f16(
                false, amat, false, bmat, (short)0, acc[j], false, false);
        }
    }

    // ---- epilogue: bias + LeakyReLU + optional skip-add, f16 NHWC store ----
#pragma unroll
    for (int j = 0; j < NT; ++j) {
        const int co = ntBase + j * 16 + (lane & 15);
        const float bval = bias[co];
#pragma unroll
        for (int r = 0; r < 8; ++r) {
            const int m = r + (hi << 3);     // C/D layout: lanes>=16 hold M=8..15
            const int p = tile * 16 + m;
            if (p < npix) {
                const int pb = p >> (2 * wShift);
                const int py = (p >> wShift) & (Hout - 1);
                const int px = p & (Hout - 1);
                const size_t idx =
                    ((((size_t)pb << wShift) + py) << wShift) + px;
                const size_t oidx = idx * (size_t)Cout + co;
                float v = acc[j][r] + bval;
                v = v >= 0.f ? v : 0.01f * v;   // LeakyReLU(0.01)
                if (skip) v += (float)skip[oidx];
                out[oidx] = (_Float16)v;
            }
        }
    }
}

// ----------------- 1x1 conv head + triangular mask + softplus --------------
__global__ void out_mask_kernel(const _Float16* __restrict__ dec0,
                                const float* __restrict__ w_out,
                                const float* __restrict__ b_out,
                                float* __restrict__ out) {
    const long total = (long)BATCH * NSZ * NSZ;
    long i = (long)blockIdx.x * 256 + threadIdx.x;
    if (i >= total) return;
    const int c = (int)(i & (NSZ - 1));
    const int r = (int)((i >> 9) & (NSZ - 1));
    float s = b_out[0];
#pragma unroll
    for (int ci = 0; ci < 16; ++ci)
        s += (float)dec0[i * 16 + ci] * w_out[ci];
    float res;
    if (r < c)       res = 0.f;                               // strict upper tri
    else if (r == c) res = (s > 20.f) ? s : log1pf(expf(s));  // softplus diag
    else             res = s;
    out[i] = res;
}

// ---------------------------------------------------------------------------
extern "C" void kernel_launch(void* const* d_in, const int* in_sizes, int n_in,
                              void* d_out, int out_size, void* d_ws, size_t ws_size,
                              hipStream_t stream) {
    (void)in_sizes; (void)n_in; (void)out_size; (void)ws_size;

    const float* xf = (const float*)d_in[0];
    const float* w_f32[13] = {
        (const float*)d_in[1],  (const float*)d_in[3],  (const float*)d_in[5],
        (const float*)d_in[7],  (const float*)d_in[9],  (const float*)d_in[11],
        (const float*)d_in[13], (const float*)d_in[15], (const float*)d_in[17],
        (const float*)d_in[19], (const float*)d_in[21], (const float*)d_in[23],
        (const float*)d_in[25] };
    const float* b_f32[13] = {
        (const float*)d_in[2],  (const float*)d_in[4],  (const float*)d_in[6],
        (const float*)d_in[8],  (const float*)d_in[10], (const float*)d_in[12],
        (const float*)d_in[14], (const float*)d_in[16], (const float*)d_in[18],
        (const float*)d_in[20], (const float*)d_in[22], (const float*)d_in[24],
        (const float*)d_in[26] };

    // per-layer (Cout, Cin) -> KREAL = 9*Cin, KPAD = round32(KREAL)
    const int cout_[12] = {16, 32, 32, 64, 64, 128, 64, 64, 32, 32, 16, 16};
    const int cin_[12]  = { 1, 16, 32, 32, 64,  64, 128, 64, 64, 32, 32, 16};

    char* ws = (char*)d_ws;
    size_t cur = 0;
    auto alloc16 = [&](size_t nelem) -> _Float16* {
        _Float16* p = (_Float16*)(ws + cur);
        cur += (nelem * 2 + 255) & ~(size_t)255;
        return p;
    };

    // transposed, K-padded f16 weights
    _Float16* wT[12];
    int kpad_[12];
    for (int i = 0; i < 12; ++i) {
        int kreal = 9 * cin_[i];
        kpad_[i]  = (kreal + 31) & ~31;
        wT[i] = alloc16((size_t)cout_[i] * kpad_[i]);
    }
    // f16 activations (NHWC)
    const size_t HW  = (size_t)NSZ * NSZ;
    _Float16* x16   = alloc16(BATCH * HW * 1);
    _Float16* enc1  = alloc16(BATCH * HW * 16);
    _Float16* down1 = alloc16(BATCH * HW / 4 * 32);
    _Float16* enc2  = alloc16(BATCH * HW / 4 * 32);
    _Float16* down2 = alloc16(BATCH * HW / 16 * 64);
    _Float16* enc3  = alloc16(BATCH * HW / 16 * 64);
    _Float16* bneck = alloc16(BATCH * HW / 64 * 128);
    _Float16* up2   = alloc16(BATCH * HW / 16 * 64);
    _Float16* dec2  = alloc16(BATCH * HW / 16 * 64);
    _Float16* up1   = alloc16(BATCH * HW / 4 * 32);
    _Float16* dec1  = alloc16(BATCH * HW / 4 * 32);
    _Float16* up0   = alloc16(BATCH * HW * 16);
    _Float16* dec0  = alloc16(BATCH * HW * 16);

    // weight prep + input cast
    for (int i = 0; i < 12; ++i) {
        int n = cout_[i] * kpad_[i];
        prep_w_kernel<<<(n + 255) / 256, 256, 0, stream>>>(
            w_f32[i], wT[i], cout_[i], 9 * cin_[i], kpad_[i]);
    }
    {
        int n = (int)(BATCH * HW);
        cast_f32_to_f16_kernel<<<(n + 255) / 256, 256, 0, stream>>>(xf, x16, n);
    }

#define CONV(CS, ST, TR, NTILE, IN, WI, SKIP, OUT, HIN_SH, WOUT_SH, COUT)         \
    do {                                                                          \
        int npix = BATCH << (2 * (WOUT_SH));                                      \
        dim3 grid((npix + 127) / 128, (COUT) / (16 * (NTILE)));                   \
        conv_wmma_kernel<CS, ST, TR, NTILE><<<grid, 256, 0, stream>>>(            \
            IN, wT[WI], b_f32[WI], SKIP, OUT, HIN_SH, WOUT_SH, COUT);             \
    } while (0)

    // Encoder
    CONV(0, 1, 0, 1, x16,   0,  nullptr, enc1,  9, 9,  16);  // enc1
    CONV(4, 2, 0, 2, enc1,  1,  nullptr, down1, 9, 8,  32);  // down1
    CONV(5, 1, 0, 2, down1, 2,  nullptr, enc2,  8, 8,  32);  // enc2
    CONV(5, 2, 0, 2, enc2,  3,  nullptr, down2, 8, 7,  64);  // down2
    CONV(6, 1, 0, 2, down2, 4,  nullptr, enc3,  7, 7,  64);  // enc3
    CONV(6, 2, 0, 2, enc3,  5,  nullptr, bneck, 7, 6, 128);  // bneck
    // Decoder (skip-add fused into epilogue)
    CONV(7, 2, 1, 1, bneck, 6,  enc3,    up2,   6, 7,  64);  // up2 + enc3
    CONV(6, 1, 0, 2, up2,   7,  nullptr, dec2,  7, 7,  64);  // dec2
    CONV(6, 2, 1, 2, dec2,  8,  enc2,    up1,   7, 8,  32);  // up1 + enc2
    CONV(5, 1, 0, 2, up1,   9,  nullptr, dec1,  8, 8,  32);  // dec1
    CONV(5, 2, 1, 1, dec1,  10, enc1,    up0,   8, 9,  16);  // up0 + enc1
    CONV(4, 1, 0, 1, up0,   11, nullptr, dec0,  9, 9,  16);  // dec0
#undef CONV

    // 1x1 head + triangular mask + softplus diagonal -> f32 output
    {
        long total = (long)BATCH * NSZ * NSZ;
        out_mask_kernel<<<(unsigned)((total + 255) / 256), 256, 0, stream>>>(
            dec0, w_f32[12], b_f32[12], (float*)d_out);
    }
}
